// SectorAwareFusionModel_15006615732284
// MI455X (gfx1250) — matrix operand order repaired and checked by reading, not compiled, not run on hardware
//
#include <hip/hip_runtime.h>

// ---------------------------------------------------------------------------
// Types for CDNA5 WMMA (wave32, v_wmma_f32_16x16x32_bf16)
// ---------------------------------------------------------------------------
typedef __attribute__((ext_vector_type(16))) __bf16 v16bf;
typedef __attribute__((ext_vector_type(8)))  float  v8f;

struct alignas(16) U4 { unsigned int a0, a1, a2, a3; };

union Frag {
    U4    q[2];
    v16bf v;
};

static __device__ __forceinline__ void frag_zero(Frag& f) {
    f.q[0] = U4{0u, 0u, 0u, 0u};
    f.q[1] = U4{0u, 0u, 0u, 0u};
}

#define BATCH 128
#define TSEQ  60
#define DMODEL 128
#define MT (BATCH * TSEQ)   // 7680 rows
#define FFN 512

// ---------------------------------------------------------------------------
// Weight packing: f32 (K,N) row-major -> bf16 (N,K) row-major (transposed)
// ---------------------------------------------------------------------------
__global__ void packT_kernel(const float* __restrict__ in, __bf16* __restrict__ out,
                             int K, int N) {
    int idx = blockIdx.x * blockDim.x + threadIdx.x;
    if (idx >= K * N) return;
    int n = idx % N, k = idx / N;
    out[(size_t)n * K + k] = (__bf16)in[idx];
}

// OIHW f32 (Co,Ci,3,3) -> bf16 [ky][kx][co][ci]
__global__ void packConv_kernel(const float* __restrict__ in, __bf16* __restrict__ out,
                                int Co, int Ci) {
    int idx = blockIdx.x * blockDim.x + threadIdx.x;
    if (idx >= Co * Ci * 9) return;
    int kx = idx % 3; int t = idx / 3;
    int ky = t % 3;  t /= 3;
    int ci = t % Ci; int co = t / Ci;
    out[(((size_t)(ky * 3 + kx)) * Co + co) * Ci + ci] = (__bf16)in[idx];
}

// ---------------------------------------------------------------------------
// Generic bf16 WMMA GEMM: out = A(M,K) @ Bt(N,K)^T + bias, optional relu.
// 16M x 64N strip per wave. K must be a multiple of 64 (true for all calls).
// Software-pipelined: ping-pong fragment buffers so loads of chunk i+1
// overlap the 4 WMMAs of chunk i (avoids s_wait_loadcnt 0 serialization).
// ---------------------------------------------------------------------------
__global__ __launch_bounds__(256)
void wmma_gemm_kernel(const __bf16* __restrict__ A, const __bf16* __restrict__ Bt,
                      const float* __restrict__ bias,
                      float* __restrict__ outF, __bf16* __restrict__ outH,
                      int M, int N, int K, int ldc, int relu) {
    int gid  = blockIdx.x * blockDim.x + threadIdx.x;
    int wave = gid >> 5;
    int lane = threadIdx.x & 31;
    int nGroups = N >> 6;
    int totalWaves = (M >> 4) * nGroups;
    if (wave >= totalWaves) return;
    int mt = wave / nGroups;
    int ng = wave % nGroups;

    int l15 = lane & 15;
    int h   = lane >> 4;
    int m   = (mt << 4) + l15;
    int klA = h << 3;        // A: K = klA..klA+7 and +16..+23
    int klB = h << 3;        // B: K = klB..klB+7 and +16..+23

    v8f acc[4] = {};
    const __bf16* aRow = A + (size_t)m * K + klA;
    const __bf16* bRow[4];
#pragma unroll
    for (int t = 0; t < 4; ++t)
        bRow[t] = Bt + (size_t)((ng << 6) + (t << 4) + l15) * K + klB;

    auto loadChunk = [&](int k0, Frag& a, Frag (&b)[4]) {
        a.q[0] = *(const U4*)(aRow + k0);
        a.q[1] = *(const U4*)(aRow + k0 + 16);
#pragma unroll
        for (int t = 0; t < 4; ++t) {
            b[t].q[0] = *(const U4*)(bRow[t] + k0);
            b[t].q[1] = *(const U4*)(bRow[t] + k0 + 16);
        }
    };

    Frag a0f, a1f;
    Frag b0f[4], b1f[4];
    loadChunk(0, a0f, b0f);

    for (int k0 = 0; k0 < K; k0 += 64) {
        loadChunk(k0 + 32, a1f, b1f);
        if (k0 + 64 < K)
            __builtin_prefetch((const void*)(aRow + k0 + 64), 0, 3);
#pragma unroll
        for (int t = 0; t < 4; ++t)
            acc[t] = __builtin_amdgcn_wmma_f32_16x16x32_bf16(
                false, a0f.v, false, b0f[t].v, (short)0, acc[t], false, false);
        if (k0 + 64 < K)
            loadChunk(k0 + 64, a0f, b0f);
#pragma unroll
        for (int t = 0; t < 4; ++t)
            acc[t] = __builtin_amdgcn_wmma_f32_16x16x32_bf16(
                false, a1f.v, false, b1f[t].v, (short)0, acc[t], false, false);
    }

    int mrow0 = (mt << 4) + (h << 3);
#pragma unroll
    for (int t = 0; t < 4; ++t) {
        int n = (ng << 6) + (t << 4) + l15;
        float bv = bias ? bias[n] : 0.0f;
#pragma unroll
        for (int j = 0; j < 8; ++j) {
            float v = acc[t][j] + bv;
            if (relu) v = fmaxf(v, 0.0f);
            size_t off = (size_t)(mrow0 + j) * ldc + n;
            if (outF) outF[off] = v;
            if (outH) outH[off] = (__bf16)v;
        }
    }
}

// ---------------------------------------------------------------------------
// Fused conv3x3(SAME) + bias + relu + maxpool2 via implicit-GEMM WMMA (NHWC).
// One wave: 2 conv rows x 16 x-positions x 16 out-channels -> 1 pooled row
// of 8 x-positions x 16 channels. 2-stage software pipeline over the
// flattened (ky,kx,kc) iteration space.
// ---------------------------------------------------------------------------
struct ConvStage { Frag b, a0, a1; };

static __device__ __forceinline__ void conv_load_stage(
    ConvStage& S, int i, int cmask, int lc,
    const __bf16* __restrict__ in, const __bf16* __restrict__ wp,
    int b, int yp, int xin, int co, int klA, int klB,
    int Cin, int Co, int H, int W) {
    int kc5 = (i & cmask) << 5;
    int kxy = i >> lc;               // ky*3 + kx
    int ky  = kxy / 3;
    int kx  = kxy - 3 * ky;
    int y0  = 2 * yp + ky - 1;
    int y1  = y0 + 1;
    int xx  = xin + kx - 1;
    bool xv = (xx >= 0) && (xx < W);

    const __bf16* wb = wp + ((size_t)kxy * Co + co) * Cin + kc5 + klB;
    S.b.q[0] = *(const U4*)(wb);
    S.b.q[1] = *(const U4*)(wb + 16);

    if ((y0 >= 0) && (y0 < H) && xv) {
        const __bf16* p = in + (((size_t)b * H + y0) * W + xx) * Cin + kc5 + klA;
        S.a0.q[0] = *(const U4*)(p);
        S.a0.q[1] = *(const U4*)(p + 16);
    } else frag_zero(S.a0);
    if ((y1 >= 0) && (y1 < H) && xv) {
        const __bf16* p = in + (((size_t)b * H + y1) * W + xx) * Cin + kc5 + klA;
        S.a1.q[0] = *(const U4*)(p);
        S.a1.q[1] = *(const U4*)(p + 16);
    } else frag_zero(S.a1);
}

__global__ __launch_bounds__(256)
void conv_wmma_kernel(const __bf16* __restrict__ in,   // (B,H,W,Cin)
                      const __bf16* __restrict__ wp,   // (3,3,Co,Cin)
                      const float* __restrict__ bias,
                      __bf16* __restrict__ out,        // (B,H/2,W/2,Co)
                      int Cin, int Co, int H, int W) {
    int gid  = blockIdx.x * blockDim.x + threadIdx.x;
    int wave = gid >> 5;
    int lane = threadIdx.x & 31;
    int PH = H >> 1, PW = W >> 1;
    int xTiles = (W + 15) >> 4;
    int coT = Co >> 4;
    int totalWaves = BATCH * PH * xTiles * coT;
    if (wave >= totalWaves) return;

    int ct = wave % coT; int r = wave / coT;
    int xg = r % xTiles; r /= xTiles;
    int yp = r % PH;     int b = r / PH;

    int l15 = lane & 15;
    int h   = lane >> 4;
    int co  = (ct << 4) + l15;
    int xin = (xg << 4) + l15;
    int klA = h << 3;
    int klB = h << 3;

    int chunks = Cin >> 5;                       // 1, 2 or 4 (power of two)
    int lc     = (chunks == 1) ? 0 : ((chunks == 2) ? 1 : 2);
    int cmask  = chunks - 1;
    int total  = 9 << lc;                        // 9 * chunks

    v8f acc0 = {}, acc1 = {};
    ConvStage S0, S1;

    conv_load_stage(S0, 0, cmask, lc, in, wp, b, yp, xin, co, klA, klB, Cin, Co, H, W);
    int i = 0;
    while (i + 2 <= total) {
        conv_load_stage(S1, i + 1, cmask, lc, in, wp, b, yp, xin, co, klA, klB, Cin, Co, H, W);
        acc0 = __builtin_amdgcn_wmma_f32_16x16x32_bf16(
            false, S0.a0.v, false, S0.b.v, (short)0, acc0, false, false);
        acc1 = __builtin_amdgcn_wmma_f32_16x16x32_bf16(
            false, S0.a1.v, false, S0.b.v, (short)0, acc1, false, false);
        if (i + 2 < total)
            conv_load_stage(S0, i + 2, cmask, lc, in, wp, b, yp, xin, co, klA, klB, Cin, Co, H, W);
        acc0 = __builtin_amdgcn_wmma_f32_16x16x32_bf16(
            false, S1.a0.v, false, S1.b.v, (short)0, acc0, false, false);
        acc1 = __builtin_amdgcn_wmma_f32_16x16x32_bf16(
            false, S1.a1.v, false, S1.b.v, (short)0, acc1, false, false);
        i += 2;
    }
    if (i < total) {  // odd tail (stage with chunks==1: 9 iterations)
        acc0 = __builtin_amdgcn_wmma_f32_16x16x32_bf16(
            false, S0.a0.v, false, S0.b.v, (short)0, acc0, false, false);
        acc1 = __builtin_amdgcn_wmma_f32_16x16x32_bf16(
            false, S0.a1.v, false, S0.b.v, (short)0, acc1, false, false);
    }

    float bv = bias[co];
#pragma unroll
    for (int j = 0; j < 8; j += 2) {
        float v00 = fmaxf(acc0[j]     + bv, 0.0f);
        float v01 = fmaxf(acc0[j + 1] + bv, 0.0f);
        float v10 = fmaxf(acc1[j]     + bv, 0.0f);
        float v11 = fmaxf(acc1[j + 1] + bv, 0.0f);
        float mx  = fmaxf(fmaxf(v00, v01), fmaxf(v10, v11));
        int xp = (xg << 3) + (j >> 1) + (h << 2);
        if (xp < PW)
            out[(((size_t)b * PH + yp) * PW + xp) * Co + co] = (__bf16)mx;
    }
}

// ---------------------------------------------------------------------------
// CNN stage 1: Cin=3 too skinny for WMMA; direct VALU conv+relu+pool.
// Output NHWC bf16 (B,112,112,32). Lanes within a wave share pixels (co fast).
// ---------------------------------------------------------------------------
__global__ void conv1_pool_kernel(const float* __restrict__ img,  // (B,3,224,224)
                                  const float* __restrict__ w,    // (32,3,3,3)
                                  const float* __restrict__ bias,
                                  __bf16* __restrict__ out) {
    int idx = blockIdx.x * blockDim.x + threadIdx.x;
    if (idx >= BATCH * 112 * 112 * 32) return;
    int co = idx & 31; int t = idx >> 5;
    int xp = t % 112; t /= 112;
    int yp = t % 112; int b = t / 112;

    float mx = 0.0f;  // relu outputs are >= 0
#pragma unroll
    for (int r = 0; r < 2; ++r)
#pragma unroll
        for (int c = 0; c < 2; ++c) {
            int y = 2 * yp + r, x = 2 * xp + c;
            float acc = bias[co];
            for (int ci = 0; ci < 3; ++ci)
#pragma unroll
                for (int ky = 0; ky < 3; ++ky)
#pragma unroll
                    for (int kx = 0; kx < 3; ++kx) {
                        int yy = y + ky - 1, xxp = x + kx - 1;
                        if (yy >= 0 && yy < 224 && xxp >= 0 && xxp < 224)
                            acc += img[(((size_t)b * 3 + ci) * 224 + yy) * 224 + xxp]
                                 * w[((co * 3 + ci) * 3 + ky) * 3 + kx];
                    }
            mx = fmaxf(mx, fmaxf(acc, 0.0f));
        }
    out[idx] = (__bf16)mx;
}

// ---------------------------------------------------------------------------
// Time-series input projection: X = ts @ in_w + in_b + pos   (f32 + bf16 out)
// ---------------------------------------------------------------------------
__global__ void ts_in_kernel(const float* __restrict__ ts, const float* __restrict__ w,
                             const float* __restrict__ bias, const float* __restrict__ pos,
                             float* __restrict__ Xf, __bf16* __restrict__ Xh) {
    int idx = blockIdx.x * blockDim.x + threadIdx.x;
    if (idx >= MT * DMODEL) return;
    int col = idx & 127; int row = idx >> 7;
    int t = row % TSEQ;
    float acc = bias[col] + pos[t * DMODEL + col];
    const float* tr = ts + (size_t)row * 25;
#pragma unroll
    for (int f = 0; f < 25; ++f)
        acc += tr[f] * w[f * DMODEL + col];
    Xf[idx] = acc;
    Xh[idx] = (__bf16)acc;
}

// ---------------------------------------------------------------------------
// Attention: one block per (b,h). K,V and softmax rows staged in LDS.
// ---------------------------------------------------------------------------
__global__ void attention_kernel(const float* __restrict__ Qf, const float* __restrict__ Kf,
                                 const float* __restrict__ Vf, __bf16* __restrict__ Oh) {
    __shared__ float ks[TSEQ * 32];
    __shared__ float vs[TSEQ * 32];
    __shared__ float ps[64 * TSEQ];
    int bh = blockIdx.x;
    int hH = bh & 3;
    int b  = bh >> 2;
    int tid = threadIdx.x;  // 64 threads
    size_t base = ((size_t)b * TSEQ) * DMODEL + hH * 32;

    for (int i = tid; i < TSEQ * 32; i += 64) {
        int s = i >> 5, d = i & 31;
        ks[i] = Kf[base + (size_t)s * DMODEL + d];
        vs[i] = Vf[base + (size_t)s * DMODEL + d];
    }
    __syncthreads();

    if (tid < TSEQ) {
        float q[32];
#pragma unroll
        for (int d = 0; d < 32; ++d)
            q[d] = Qf[base + (size_t)tid * DMODEL + d] * 0.17677669529663687f;
        float mx = -1e30f;
        for (int s = 0; s < TSEQ; ++s) {
            float sc = 0.0f;
#pragma unroll
            for (int d = 0; d < 32; ++d) sc += q[d] * ks[s * 32 + d];
            ps[tid * TSEQ + s] = sc;
            mx = fmaxf(mx, sc);
        }
        float sum = 0.0f;
        for (int s = 0; s < TSEQ; ++s) {
            float e = __expf(ps[tid * TSEQ + s] - mx);
            ps[tid * TSEQ + s] = e;
            sum += e;
        }
        float inv = 1.0f / sum;
        float o[32];
#pragma unroll
        for (int d = 0; d < 32; ++d) o[d] = 0.0f;
        for (int s = 0; s < TSEQ; ++s) {
            float p = ps[tid * TSEQ + s] * inv;
#pragma unroll
            for (int d = 0; d < 32; ++d) o[d] += p * vs[s * 32 + d];
        }
#pragma unroll
        for (int d = 0; d < 32; ++d)
            Oh[base + (size_t)tid * DMODEL + d] = (__bf16)o[d];
    }
}

// ---------------------------------------------------------------------------
// Residual + LayerNorm (one wave per 128-elem row; shfl_xor wave32 reduction)
// ---------------------------------------------------------------------------
__global__ void ln_kernel(const float* __restrict__ x, const float* __restrict__ r,
                          const float* __restrict__ g, const float* __restrict__ bta,
                          float* __restrict__ outF, __bf16* __restrict__ outH, int Mrows) {
    int gid  = blockIdx.x * blockDim.x + threadIdx.x;
    int wave = gid >> 5;
    int lane = threadIdx.x & 31;
    if (wave >= Mrows) return;
    size_t row = (size_t)wave * DMODEL;
    float v[4], s = 0.0f, ss = 0.0f;
#pragma unroll
    for (int j = 0; j < 4; ++j) {
        float a = x[row + lane * 4 + j] + r[row + lane * 4 + j];
        v[j] = a; s += a; ss += a * a;
    }
#pragma unroll
    for (int m = 16; m >= 1; m >>= 1) {
        s  += __shfl_xor(s,  m, 32);
        ss += __shfl_xor(ss, m, 32);
    }
    float mean = s * (1.0f / 128.0f);
    float var  = ss * (1.0f / 128.0f) - mean * mean;
    float rs   = rsqrtf(var + 1e-5f);
#pragma unroll
    for (int j = 0; j < 4; ++j) {
        int c = lane * 4 + j;
        float y = (v[j] - mean) * rs * g[c] + bta[c];
        outF[row + c] = y;
        outH[row + c] = (__bf16)y;
    }
}

// Mean over T -> fused[:, 0:128] (bf16)
__global__ void meant_kernel(const float* __restrict__ Xf, __bf16* __restrict__ fused) {
    int idx = blockIdx.x * blockDim.x + threadIdx.x;
    if (idx >= BATCH * DMODEL) return;
    int d = idx & 127, b = idx >> 7;
    float s = 0.0f;
    for (int t = 0; t < TSEQ; ++t)
        s += Xf[((size_t)b * TSEQ + t) * DMODEL + d];
    fused[(size_t)b * 256 + d] = (__bf16)(s * (1.0f / TSEQ));
}

// Global average pool over 14x14 -> gapH (B,256) bf16
__global__ void gap_kernel(const __bf16* __restrict__ p4, __bf16* __restrict__ gapH) {
    int idx = blockIdx.x * blockDim.x + threadIdx.x;
    if (idx >= BATCH * 256) return;
    int co = idx & 255, b = idx >> 8;
    float s = 0.0f;
    for (int i = 0; i < 196; ++i)
        s += (float)p4[((size_t)b * 196 + i) * 256 + co];
    gapH[idx] = (__bf16)(s / 196.0f);
}

// ---------------------------------------------------------------------------
// Sector heads: per-sample gathered MLP + softmax. One block per sample.
// ---------------------------------------------------------------------------
__global__ void heads_kernel(const float* __restrict__ shared_f, const int* __restrict__ sid,
                             const float* __restrict__ w1, const float* __restrict__ b1,
                             const float* __restrict__ w2, const float* __restrict__ b2,
                             float* __restrict__ outLogits, float* __restrict__ outProbs) {
    __shared__ float hs[64];
    __shared__ float lg[3];
    int b = blockIdx.x, t = threadIdx.x;  // 64 threads
    int s = sid[b];
    const float* W1 = w1 + (size_t)s * 128 * 64;
    const float* sh = shared_f + (size_t)b * 128;
    float acc = b1[s * 64 + t];
    for (int d = 0; d < 128; ++d)
        acc += sh[d] * W1[d * 64 + t];
    hs[t] = fmaxf(acc, 0.0f);
    __syncthreads();
    if (t < 3) {
        const float* W2 = w2 + (size_t)s * 64 * 3;
        float a = b2[s * 3 + t];
        for (int k = 0; k < 64; ++k)
            a += hs[k] * W2[k * 3 + t];
        outLogits[b * 3 + t] = a;
        lg[t] = a;
    }
    __syncthreads();
    if (t < 3) {
        float m = fmaxf(lg[0], fmaxf(lg[1], lg[2]));
        float e0 = __expf(lg[0] - m), e1 = __expf(lg[1] - m), e2 = __expf(lg[2] - m);
        outProbs[b * 3 + t] = __expf(lg[t] - m) / (e0 + e1 + e2);
    }
}

// ---------------------------------------------------------------------------
// Host orchestration
// ---------------------------------------------------------------------------
extern "C" void kernel_launch(void* const* d_in, const int* in_sizes, int n_in,
                              void* d_out, int out_size, void* d_ws, size_t ws_size,
                              hipStream_t stream) {
    (void)in_sizes; (void)n_in; (void)out_size; (void)ws_size;

    const float* ts  = (const float*)d_in[0];
    const float* img = (const float*)d_in[1];
    const int*   sid = (const int*)d_in[2];
    // params flattened: cnn(convs w/b x4, fc_b=11, fc_w=12), fusion(b=13,w=14),
    // heads(b1=15,b2=16,w1=17,w2=18), ts(in_b=19,in_w=20, layers 21..84, pos=85)

    char* ws = (char*)d_ws;
    size_t cur = 0;
    auto alloc = [&](size_t bytes) -> char* {
        char* p = ws + cur;
        cur = (cur + bytes + 255) & ~(size_t)255;
        return p;
    };

    float*  Xf = (float*) alloc((size_t)MT * 128 * 4);
    __bf16* Xh = (__bf16*)alloc((size_t)MT * 128 * 2);
    float*  Qf = (float*) alloc((size_t)MT * 128 * 4);
    float*  Kf = (float*) alloc((size_t)MT * 128 * 4);
    float*  Vf = (float*) alloc((size_t)MT * 128 * 4);
    __bf16* Oh = (__bf16*)alloc((size_t)MT * 128 * 2);
    float*  Pf = (float*) alloc((size_t)MT * 128 * 4);
    __bf16* Hh = (__bf16*)alloc((size_t)MT * 512 * 2);

    __bf16* wT[4][6];
    for (int l = 0; l < 4; ++l) {
        wT[l][0] = (__bf16*)alloc(128 * 128 * 2);  // wqT
        wT[l][1] = (__bf16*)alloc(128 * 128 * 2);  // wkT
        wT[l][2] = (__bf16*)alloc(128 * 128 * 2);  // wvT
        wT[l][3] = (__bf16*)alloc(128 * 128 * 2);  // woT
        wT[l][4] = (__bf16*)alloc(512 * 128 * 2);  // w1T (N=512,K=128)
        wT[l][5] = (__bf16*)alloc(128 * 512 * 2);  // w2T (N=128,K=512)
    }
    __bf16* cw2 = (__bf16*)alloc(9 * 64 * 32 * 2);
    __bf16* cw3 = (__bf16*)alloc(9 * 128 * 64 * 2);
    __bf16* cw4 = (__bf16*)alloc(9 * 256 * 128 * 2);
    __bf16* fcT  = (__bf16*)alloc(128 * 256 * 2);
    __bf16* fusT = (__bf16*)alloc(128 * 256 * 2);

    __bf16* p1 = (__bf16*)alloc((size_t)BATCH * 112 * 112 * 32 * 2);
    __bf16* p2 = (__bf16*)alloc((size_t)BATCH * 56 * 56 * 64 * 2);
    __bf16* p3 = (__bf16*)alloc((size_t)BATCH * 28 * 28 * 128 * 2);
    __bf16* p4 = (__bf16*)alloc((size_t)BATCH * 14 * 14 * 256 * 2);
    __bf16* gapH  = (__bf16*)alloc(BATCH * 256 * 2);
    __bf16* fused = (__bf16*)alloc(BATCH * 256 * 2);

    auto packT = [&](const float* in, __bf16* out, int K, int N) {
        int n = K * N;
        packT_kernel<<<(n + 255) / 256, 256, 0, stream>>>(in, out, K, N);
    };
    auto gemm = [&](const __bf16* A, const __bf16* Bt, const float* bias,
                    float* oF, __bf16* oH, int M, int N, int K, int ldc, int relu) {
        int waves = (M / 16) * (N / 64);
        int thr = waves * 32;
        wmma_gemm_kernel<<<(thr + 255) / 256, 256, 0, stream>>>(
            A, Bt, bias, oF, oH, M, N, K, ldc, relu);
    };

    // ---- weight packing ----
    for (int l = 0; l < 4; ++l) {
        int base = 21 + 16 * l;
        packT((const float*)d_in[base + 14], wT[l][0], 128, 128);  // wq
        packT((const float*)d_in[base + 12], wT[l][1], 128, 128);  // wk
        packT((const float*)d_in[base + 15], wT[l][2], 128, 128);  // wv
        packT((const float*)d_in[base + 13], wT[l][3], 128, 128);  // wo
        packT((const float*)d_in[base + 10], wT[l][4], 128, 512);  // w1
        packT((const float*)d_in[base + 11], wT[l][5], 512, 128);  // w2
    }
    packT((const float*)d_in[12], fcT,  256, 128);
    packT((const float*)d_in[14], fusT, 256, 128);
    {
        int n2 = 64 * 32 * 9, n3 = 128 * 64 * 9, n4 = 256 * 128 * 9;
        packConv_kernel<<<(n2 + 255) / 256, 256, 0, stream>>>((const float*)d_in[5], cw2, 64, 32);
        packConv_kernel<<<(n3 + 255) / 256, 256, 0, stream>>>((const float*)d_in[7], cw3, 128, 64);
        packConv_kernel<<<(n4 + 255) / 256, 256, 0, stream>>>((const float*)d_in[9], cw4, 256, 128);
    }

    // ---- time-series branch ----
    ts_in_kernel<<<(MT * 128 + 255) / 256, 256, 0, stream>>>(
        ts, (const float*)d_in[20], (const float*)d_in[19], (const float*)d_in[85], Xf, Xh);

    for (int l = 0; l < 4; ++l) {
        int base = 21 + 16 * l;
        const float* b1  = (const float*)d_in[base + 0];
        const float* b2  = (const float*)d_in[base + 1];
        const float* bk  = (const float*)d_in[base + 2];
        const float* bo  = (const float*)d_in[base + 3];
        const float* bq  = (const float*)d_in[base + 4];
        const float* bv  = (const float*)d_in[base + 5];
        const float* l1b = (const float*)d_in[base + 6];
        const float* l1g = (const float*)d_in[base + 7];
        const float* l2b = (const float*)d_in[base + 8];
        const float* l2g = (const float*)d_in[base + 9];

        gemm(Xh, wT[l][0], bq, Qf, nullptr, MT, 128, 128, 128, 0);
        gemm(Xh, wT[l][1], bk, Kf, nullptr, MT, 128, 128, 128, 0);
        gemm(Xh, wT[l][2], bv, Vf, nullptr, MT, 128, 128, 128, 0);
        attention_kernel<<<BATCH * 4, 64, 0, stream>>>(Qf, Kf, Vf, Oh);
        gemm(Oh, wT[l][3], bo, Pf, nullptr, MT, 128, 128, 128, 0);
        ln_kernel<<<(MT * 32 + 255) / 256, 256, 0, stream>>>(Xf, Pf, l1g, l1b, Xf, Xh, MT);
        gemm(Xh, wT[l][4], b1, nullptr, Hh, MT, 512, 128, 512, 1);
        gemm(Hh, wT[l][5], b2, Pf, nullptr, MT, 128, 512, 128, 0);
        ln_kernel<<<(MT * 32 + 255) / 256, 256, 0, stream>>>(Xf, Pf, l2g, l2b, Xf, Xh, MT);
    }
    meant_kernel<<<(BATCH * 128 + 255) / 256, 256, 0, stream>>>(Xf, fused);

    // ---- CNN branch ----
    {
        int n1 = BATCH * 112 * 112 * 32;
        conv1_pool_kernel<<<(n1 + 255) / 256, 256, 0, stream>>>(
            img, (const float*)d_in[3], (const float*)d_in[4], p1);
    }
    auto convw = [&](const __bf16* in, const __bf16* wp, const float* bias, __bf16* out,
                     int Cin, int Co, int H, int W) {
        int PH = H / 2, xT = (W + 15) / 16, cT = Co / 16;
        long waves = (long)BATCH * PH * xT * cT;
        int thr = (int)(waves * 32);
        conv_wmma_kernel<<<(thr + 255) / 256, 256, 0, stream>>>(in, wp, bias, out, Cin, Co, H, W);
    };
    convw(p1, cw2, (const float*)d_in[6],  p2, 32,  64,  112, 112);
    convw(p2, cw3, (const float*)d_in[8],  p3, 64,  128, 56,  56);
    convw(p3, cw4, (const float*)d_in[10], p4, 128, 256, 28,  28);

    gap_kernel<<<(BATCH * 256 + 255) / 256, 256, 0, stream>>>(p4, gapH);
    // fc: (B,256) @ (256,128) -> bf16 into fused[:,128:256] (ldc = 256)
    gemm(gapH, fcT, (const float*)d_in[11], nullptr, fused + 128, BATCH, 128, 256, 256, 0);

    // ---- fusion: shared = relu(fused @ W + b), f32 directly into d_out ----
    float* out_logits = (float*)d_out;
    float* out_probs  = (float*)d_out + BATCH * 3;
    float* out_shared = (float*)d_out + BATCH * 6;
    gemm(fused, fusT, (const float*)d_in[13], out_shared, nullptr, BATCH, 128, 256, 128, 1);

    // ---- sector heads ----
    heads_kernel<<<BATCH, 64, 0, stream>>>(
        out_shared, sid, (const float*)d_in[17], (const float*)d_in[15],
        (const float*)d_in[18], (const float*)d_in[16], out_logits, out_probs);
}
